// GIN_22316650070138
// MI455X (gfx1250) — matrix operand order repaired and checked by reading, not compiled, not run on hardware
//
#include <hip/hip_runtime.h>
#include <stdint.h>

// ---------------------------------------------------------------------------
// Types for CDNA5 WMMA (wave32)
// ---------------------------------------------------------------------------
typedef __attribute__((ext_vector_type(16))) __bf16          v16bf;
typedef __attribute__((ext_vector_type(16))) unsigned short  v16us;
typedef __attribute__((ext_vector_type(8)))  float           v8f;

union BfVec { v16bf bf; v16us us; };

__device__ __forceinline__ unsigned short f2bf_rne(float f) {
  unsigned int u = __float_as_uint(f);
  u += 0x7FFFu + ((u >> 16) & 1u);          // round-to-nearest-even
  return (unsigned short)(u >> 16);
}
__device__ __forceinline__ float bf2f(unsigned short b) {
  return __uint_as_float(((unsigned int)b) << 16);
}

#define HIDC 128   // feature width (D_IN == HID == 128)

// ---------------------------------------------------------------------------
// Pack 4 weight matrices (128x128 f32, row-major [k][n]) into the wave32
// WMMA B-operand layout for 16x16x32 bf16, split into hi/lo bf16 parts.
// Packed element index: r = ((kstep*8 + ntile)*32 + lane)*16 + j
//   k = kstep*32 + (j>>3)*16 + (lane>>4)*8 + (j&7)   (lanes 0-15: K 0-7,16-23; lanes 16-31: K 8-15,24-31)
//   n = ntile*16 + (lane&15)
// Per matrix: 16384 hi bf16 followed by 16384 lo bf16 (32768 ushort total).
// ---------------------------------------------------------------------------
__global__ __launch_bounds__(256) void gin_pack_weights(
    const float* __restrict__ W0, const float* __restrict__ W1,
    const float* __restrict__ W2, const float* __restrict__ W3,
    unsigned short* __restrict__ Bpk)
{
  int i = blockIdx.x * 256 + threadIdx.x;        // 0 .. 65535
  int m = i >> 14;
  int r = i & 16383;
  const float* W = (m == 0) ? W0 : (m == 1) ? W1 : (m == 2) ? W2 : W3;
  int j    = r & 15;
  int lane = (r >> 4) & 31;
  int nt   = (r >> 9) & 7;
  int ks   = (r >> 12) & 3;
  int k = ks * 32 + ((j >> 3) << 4) + ((lane >> 4) << 3) + (j & 7);
  int n = nt * 16 + (lane & 15);
  float f = W[k * HIDC + n];
  unsigned short hi = f2bf_rne(f);
  unsigned short lo = f2bf_rne(f - bf2f(hi));
  unsigned short* base = Bpk + (size_t)m * 32768;
  base[r]         = hi;
  base[16384 + r] = lo;
}

// ---------------------------------------------------------------------------
// Edge scatter:  Agg[dst] += X[src]   (one lane per (edge, float4))
// ---------------------------------------------------------------------------
__global__ __launch_bounds__(256) void gin_scatter_add(
    const float* __restrict__ X, const int* __restrict__ src,
    const int* __restrict__ dst, float* __restrict__ Agg, int nEdges)
{
  int tid = blockIdx.x * 256 + threadIdx.x;
  int e = tid >> 5;
  if (e >= nEdges) return;
  int q = (tid & 31) << 2;
  int s = src[e], d = dst[e];
  float4 v = *(const float4*)(X + (size_t)s * HIDC + q);
  float* p = Agg + (size_t)d * HIDC + q;
  atomicAdd(p + 0, v.x);
  atomicAdd(p + 1, v.y);
  atomicAdd(p + 2, v.z);
  atomicAdd(p + 3, v.w);
}

// ---------------------------------------------------------------------------
// WMMA GEMM:  Out = act( (A0 [+ A1]) @ W + bias )
//   A: nrows x 128 f32 (row-major), W packed bf16 hi/lo (see gin_pack_weights)
//   block = 256 threads (8 waves), M-tile = 128 rows, wave -> 16-row strip.
//   bf16x3: acc += Ah*Bh + Al*Bh + Ah*Bl  (f32 accumulate) -> ~fp32 accuracy.
//   Non-DUAL A staging uses gfx1250 async global->LDS DMA (ASYNCcnt).
//   In-place safe: the block's A rows are fully staged to LDS before stores.
// ---------------------------------------------------------------------------
template<bool DUAL, bool RELU>
__global__ __launch_bounds__(256, 2) void gin_gemm128(
    const float* __restrict__ A0, const float* __restrict__ A1,
    const unsigned short* __restrict__ Bpk, const float* __restrict__ bias,
    float* __restrict__ Out, int nrows)
{
  __shared__ float As[128 * 132];                 // pad to 132 floats/row
  const int t  = threadIdx.x;
  const int r0 = blockIdx.x * 128;

  // ---- stage A tile, fusing A0 + A1 when DUAL; async DMA otherwise ----
  #pragma unroll
  for (int i = 0; i < 16; ++i) {
    int elem = i * 1024 + t * 4;
    int row = elem >> 7, col = elem & 127;
    int gr = r0 + row;
    float* dstp = &As[row * 132 + col];
    if (DUAL) {
      float4 v = make_float4(0.f, 0.f, 0.f, 0.f);
      if (gr < nrows) {
        v = *(const float4*)(A0 + (size_t)gr * HIDC + col);
        float4 u = *(const float4*)(A1 + (size_t)gr * HIDC + col);
        v.x += u.x; v.y += u.y; v.z += u.z; v.w += u.w;
      }
      *(float4*)dstp = v;
    } else {
      if (gr < nrows) {
        // global -> LDS async DMA (GVS mode: SGPR base + 32-bit VGPR offset)
        unsigned ldsoff = (unsigned)(uintptr_t)dstp;      // LDS byte offset
        unsigned goff   = (unsigned)gr * (HIDC * 4u) + (unsigned)col * 4u;
        asm volatile("global_load_async_to_lds_b128 %0, %1, %2"
                     :: "v"(ldsoff), "v"(goff), "s"(A0)
                     : "memory");
      } else {
        *(float4*)dstp = make_float4(0.f, 0.f, 0.f, 0.f);
      }
    }
  }
  if (!DUAL) {
    asm volatile("s_wait_asynccnt 0" ::: "memory");
  }
  __syncthreads();

  const int wave = t >> 5;
  const int lane = t & 31;
  const int mrow  = lane & 15;
  const int khalf = (lane >> 4) << 3;             // A layout: 0 or 8

  v8f acc[8];
  #pragma unroll
  for (int nt = 0; nt < 8; ++nt) { v8f z = {}; acc[nt] = z; }

  #pragma unroll
  for (int ks = 0; ks < 4; ++ks) {
    // ---- build A operand (16x32 bf16 layout), hi/lo split ----
    const float* ap = &As[(wave * 16 + mrow) * 132 + ks * 32 + khalf];
    float4 p0 = *(const float4*)(ap);
    float4 p1 = *(const float4*)(ap + 4);
    float4 p2 = *(const float4*)(ap + 16);
    float4 p3 = *(const float4*)(ap + 20);
    float fa[16] = { p0.x, p0.y, p0.z, p0.w,  p1.x, p1.y, p1.z, p1.w,
                     p2.x, p2.y, p2.z, p2.w,  p3.x, p3.y, p3.z, p3.w };
    BfVec ah, al;
    #pragma unroll
    for (int j = 0; j < 16; ++j) {
      unsigned short h = f2bf_rne(fa[j]);
      ah.us[j] = h;
      al.us[j] = f2bf_rne(fa[j] - bf2f(h));
    }

    // ---- 8 N-tiles, 3 WMMAs each ----
    #pragma unroll
    for (int nt = 0; nt < 8; ++nt) {
      const v16us* bp = (const v16us*)Bpk + (((ks * 8 + nt) * 32) + lane);
      BfVec bh, bl;
      bh.us = bp[0];
      bl.us = bp[1024];                            // lo part: +16384 ushort
      acc[nt] = __builtin_amdgcn_wmma_f32_16x16x32_bf16(
          false, ah.bf, false, bh.bf, (short)0, acc[nt], false, false);
      acc[nt] = __builtin_amdgcn_wmma_f32_16x16x32_bf16(
          false, al.bf, false, bh.bf, (short)0, acc[nt], false, false);
      acc[nt] = __builtin_amdgcn_wmma_f32_16x16x32_bf16(
          false, ah.bf, false, bl.bf, (short)0, acc[nt], false, false);
    }
  }

  // ---- epilogue: bias (+ReLU), store f32 ----
  const int nlo   = lane & 15;
  const int rbase = r0 + wave * 16 + ((lane >> 4) << 3);
  #pragma unroll
  for (int nt = 0; nt < 8; ++nt) {
    int n = nt * 16 + nlo;
    float b = bias[n];
    #pragma unroll
    for (int v = 0; v < 8; ++v) {
      int gr = rbase + v;
      if (gr < nrows) {
        float o = acc[nt][v] + b;
        if (RELU) o = fmaxf(o, 0.f);
        Out[(size_t)gr * HIDC + n] = o;
      }
    }
  }
}

// ---------------------------------------------------------------------------
// Global mean-pool accumulation: sums[batch[n]] += H[n], cnts[batch[n]] += 1
// ---------------------------------------------------------------------------
__global__ __launch_bounds__(256) void gin_pool_add(
    const float* __restrict__ H, const int* __restrict__ batch,
    float* __restrict__ sums, float* __restrict__ cnts, int nNodes)
{
  int tid = blockIdx.x * 256 + threadIdx.x;
  int n = tid >> 5;
  if (n >= nNodes) return;
  int q = (tid & 31) << 2;
  int g = batch[n];
  float4 v = *(const float4*)(H + (size_t)n * HIDC + q);
  float* p = sums + (size_t)g * HIDC + q;
  atomicAdd(p + 0, v.x);
  atomicAdd(p + 1, v.y);
  atomicAdd(p + 2, v.z);
  atomicAdd(p + 3, v.w);
  if (q == 0) atomicAdd(&cnts[g], 1.0f);
}

// ---------------------------------------------------------------------------
// Head:  out[g][o] = (sums[g]/max(cnt,1)) @ Wlin + blin
// ---------------------------------------------------------------------------
__global__ __launch_bounds__(256) void gin_head(
    const float* __restrict__ sums, const float* __restrict__ cnts,
    const float* __restrict__ Wlin, const float* __restrict__ blin,
    float* __restrict__ out, int nGraphs)
{
  int tid = blockIdx.x * 256 + threadIdx.x;
  if (tid >= nGraphs * 6) return;
  int g = tid / 6, o = tid - g * 6;
  float inv = 1.0f / fmaxf(cnts[g], 1.0f);
  float acc = blin[o];
  const float* s = sums + (size_t)g * HIDC;
  #pragma unroll 8
  for (int k = 0; k < HIDC; ++k) acc += s[k] * inv * Wlin[k * 6 + o];
  out[tid] = acc;
}

// ---------------------------------------------------------------------------
// Launch
// ---------------------------------------------------------------------------
extern "C" void kernel_launch(void* const* d_in, const int* in_sizes, int n_in,
                              void* d_out, int out_size, void* d_ws, size_t ws_size,
                              hipStream_t stream) {
  const float* x     = (const float*)d_in[0];
  const int*   ei    = (const int*)d_in[1];
  const int*   batch = (const int*)d_in[2];
  const float* W1a   = (const float*)d_in[3];
  const float* b1a   = (const float*)d_in[4];
  const float* W1b   = (const float*)d_in[5];
  const float* b1b   = (const float*)d_in[6];
  const float* W2a   = (const float*)d_in[7];
  const float* b2a   = (const float*)d_in[8];
  const float* W2b   = (const float*)d_in[9];
  const float* b2b   = (const float*)d_in[10];
  const float* Wlin  = (const float*)d_in[11];
  const float* blin  = (const float*)d_in[12];
  float* out = (float*)d_out;

  const int nNodes  = in_sizes[0] / HIDC;
  const int nEdges  = in_sizes[1] / 2;
  const int nGraphs = out_size / 6;
  const int* esrc = ei;
  const int* edst = ei + nEdges;

  // workspace layout
  const size_t nodeBytes = (size_t)nNodes * HIDC * sizeof(float);
  char* ws = (char*)d_ws;
  float*          bufA = (float*)ws;                         // agg/t scratch
  float*          bufB = (float*)(ws + nodeBytes);           // h1 / h2
  unsigned short* Bpk  = (unsigned short*)(ws + 2 * nodeBytes); // 4*32768 u16
  float*          sums = (float*)(ws + 2 * nodeBytes + 262144);
  float*          cnts = (float*)(ws + 2 * nodeBytes + 262144 +
                                  (size_t)nGraphs * HIDC * sizeof(float));
  unsigned short* B1a = Bpk;
  unsigned short* B1b = Bpk + 32768;
  unsigned short* B2a = Bpk + 65536;
  unsigned short* B2b = Bpk + 98304;

  const int gemmGrid    = (nNodes + 127) / 128;
  const int scatterGrid = (nEdges * 32 + 255) / 256;
  const int poolGrid    = (nNodes * 32 + 255) / 256;

  // weight packing (tiny, redone each call for determinism)
  gin_pack_weights<<<256, 256, 0, stream>>>(W1a, W1b, W2a, W2b, Bpk);

  // ---- layer 1 ----
  hipMemsetAsync(bufA, 0, nodeBytes, stream);
  gin_scatter_add<<<scatterGrid, 256, 0, stream>>>(x, esrc, edst, bufA, nEdges);
  // t1 = relu((x + agg1) @ W1a + b1a)   (in-place into bufA)
  gin_gemm128<true,  true ><<<gemmGrid, 256, 0, stream>>>(x, bufA, B1a, b1a, bufA, nNodes);
  // h1 = relu(t1 @ W1b + b1b)           (outer conv ReLU fused)
  gin_gemm128<false, true ><<<gemmGrid, 256, 0, stream>>>(bufA, nullptr, B1b, b1b, bufB, nNodes);

  // ---- layer 2 ----
  hipMemsetAsync(bufA, 0, nodeBytes, stream);
  gin_scatter_add<<<scatterGrid, 256, 0, stream>>>(bufB, esrc, edst, bufA, nEdges);
  // t2 = relu((h1 + agg2) @ W2a + b2a)  (in-place into bufA)
  gin_gemm128<true,  true ><<<gemmGrid, 256, 0, stream>>>(bufB, bufA, B2a, b2a, bufA, nNodes);
  // h2 = t2 @ W2b + b2b                 (no ReLU)
  gin_gemm128<false, false><<<gemmGrid, 256, 0, stream>>>(bufA, nullptr, B2b, b2b, bufB, nNodes);

  // ---- mean pool + head ----
  hipMemsetAsync(sums, 0, (size_t)nGraphs * HIDC * sizeof(float), stream);
  hipMemsetAsync(cnts, 0, (size_t)nGraphs * sizeof(float), stream);
  gin_pool_add<<<poolGrid, 256, 0, stream>>>(bufB, batch, sums, cnts, nNodes);
  gin_head<<<(nGraphs * 6 + 255) / 256, 256, 0, stream>>>(sums, cnts, Wlin, blin, out, nGraphs);
}
